// GQAttention_13503377178848
// MI455X (gfx1250) — matrix operand order repaired and checked by reading, not compiled, not run on hardware
//
#include <hip/hip_runtime.h>
#include <hip/hip_bf16.h>

// ---------------------------------------------------------------------------
// Problem constants (from reference)
// ---------------------------------------------------------------------------
constexpr int Bc   = 4;
constexpr int Lc   = 2048;
constexpr int Dc   = 2048;
constexpr int Hc   = 16;
constexpr int KVHc = 4;
constexpr int HDc  = 128;
constexpr int Mc   = 16;          // rope-free prefix length
constexpr int BLc  = Bc * Lc;     // 8192 rows
constexpr int KVD  = KVHc * HDc;  // 512
constexpr float EPSc = 1.1920929e-07f;

typedef __bf16 bf16_t;
typedef __attribute__((ext_vector_type(16))) __bf16 v16bf;
typedef __attribute__((ext_vector_type(8)))  __bf16 v8bf;
typedef __attribute__((ext_vector_type(8)))  float  v8f;
typedef __attribute__((ext_vector_type(4)))  float  v4f;

__device__ __forceinline__ v8f zero_v8f() {
    v8f r;
#pragma unroll
    for (int i = 0; i < 8; ++i) r[i] = 0.0f;
    return r;
}

__device__ __forceinline__ v16bf cat8(v8bf a, v8bf b) {
    return __builtin_shufflevector(a, b, 0,1,2,3,4,5,6,7,8,9,10,11,12,13,14,15);
}

__device__ __forceinline__ v8f wmma_bf16(v16bf a, v16bf b, v8f c) {
    return __builtin_amdgcn_wmma_f32_16x16x32_bf16(false, a, false, b,
                                                   (short)0, c, false, false);
}

// CDNA5 async copy: global -> LDS, 16 bytes per lane, tracked by ASYNCcnt.
__device__ __forceinline__ void async_g2l_b128(unsigned lds_addr, const void* gptr) {
    unsigned long long ga = (unsigned long long)(uintptr_t)gptr;
    asm volatile("global_load_async_to_lds_b128 %0, %1, off"
                 :: "v"(lds_addr), "v"(ga) : "memory");
}
__device__ __forceinline__ void wait_asynccnt0() {
    asm volatile("s_wait_asynccnt 0x0" ::: "memory");
}
__device__ __forceinline__ unsigned lds_off(const void* p) {
    // Flat shared pointers map to LDS with the offset in the low 32 bits
    return (unsigned)(uintptr_t)p;
}

// ---------------------------------------------------------------------------
// Kernel 1: f32 -> bf16 cast (8 elements / thread)
// ---------------------------------------------------------------------------
__global__ __launch_bounds__(256)
void cast_f32_bf16(const float* __restrict__ in, bf16_t* __restrict__ out, int n8) {
    int i = blockIdx.x * blockDim.x + threadIdx.x;
    if (i >= n8) return;
    const v4f* p = reinterpret_cast<const v4f*>(in) + (size_t)i * 2;
    v4f a = p[0], b = p[1];
    v8bf o;
#pragma unroll
    for (int k = 0; k < 4; ++k) { o[k] = (bf16_t)a[k]; o[4 + k] = (bf16_t)b[k]; }
    reinterpret_cast<v8bf*>(out)[i] = o;
}

// ---------------------------------------------------------------------------
// Kernel 2: bf16 GEMM  C[M x N] = A[M x K] * Bw[N x K]^T  via WMMA.
// Block = 256 threads computes a 128x128 tile. Per 32-wide k-slab the block
// async-copies A(128x32) and B(128x32) into double-buffered LDS
// (global_load_async_to_lds_b128 / s_wait_asynccnt), then each of the 8 waves
// computes a 32(M)x64(N) sub-tile: 8 WMMAs per k-slab from LDS fragments.
// Epilogue scatter modes:
//   0: f32 row-major (final projection, writes d_out)
//   1: bf16 Q layout  (b, H,  l, hd)
//   2: bf16 K layout  (b, KVH,l, hd)
//   3: bf16 V^T layout(b, KVH,hd, l)
// ---------------------------------------------------------------------------
__global__ __launch_bounds__(256)
void gemm_bf16(const bf16_t* __restrict__ A, const bf16_t* __restrict__ Bw,
               void* __restrict__ Cout, int Msz, int Nsz, int Ksz, int mode) {
    __shared__ __align__(16) bf16_t Atile[2][128 * 32];   // 8 KB per buffer
    __shared__ __align__(16) bf16_t Btile[2][128 * 32];   // 8 KB per buffer

    const int tid  = threadIdx.x;
    const int wib  = tid >> 5;
    const int lane = tid & 31;
    const int ln = lane & 15, hw = lane >> 4;
    const int mw = wib & 3;        // 4 M sub-tiles of 32 rows
    const int nw = wib >> 2;       // 2 N sub-tiles of 64 cols
    const int nblk = Nsz / 128;
    const int m0b = (blockIdx.x / nblk) * 128;
    const int n0b = (blockIdx.x % nblk) * 128;

    // 256 threads stage 128x32 bf16 (8 KB) in 2 rounds of 16B chunks.
    auto stage = [&](int k0, int buf) {
#pragma unroll
        for (int rnd = 0; rnd < 2; ++rnd) {
            const int c   = tid + rnd * 256;      // chunk id, 4 chunks per row
            const int row = c >> 2;
            const int kc  = (c & 3) * 8;
            async_g2l_b128(lds_off(&Atile[buf][row * 32 + kc]),
                           A + (size_t)(m0b + row) * Ksz + k0 + kc);
            async_g2l_b128(lds_off(&Btile[buf][row * 32 + kc]),
                           Bw + (size_t)(n0b + row) * Ksz + k0 + kc);
        }
    };

    v8f acc[2][4];
#pragma unroll
    for (int rh = 0; rh < 2; ++rh)
#pragma unroll
        for (int nt = 0; nt < 4; ++nt) acc[rh][nt] = zero_v8f();

    stage(0, 0);
    wait_asynccnt0();
    __syncthreads();

    int buf = 0;
    for (int k0 = 0; k0 < Ksz; k0 += 32) {
        if (k0 + 32 < Ksz) stage(k0 + 32, buf ^ 1);

        // A fragments: rows mw*32 + rh*16 + ln, K halves per ISA A-16x32 layout
        v16bf af[2];
#pragma unroll
        for (int rh = 0; rh < 2; ++rh) {
            const bf16_t* ap = &Atile[buf][(mw * 32 + rh * 16 + ln) * 32 + hw * 8];
            af[rh] = cat8(*(const v8bf*)ap, *(const v8bf*)(ap + 16));
        }
#pragma unroll
        for (int nt = 0; nt < 4; ++nt) {
            const bf16_t* bp = &Btile[buf][(nw * 64 + nt * 16 + ln) * 32 + hw * 8];
            v16bf bfr = cat8(*(const v8bf*)bp, *(const v8bf*)(bp + 16));
            acc[0][nt] = wmma_bf16(af[0], bfr, acc[0][nt]);
            acc[1][nt] = wmma_bf16(af[1], bfr, acc[1][nt]);
        }

        if (k0 + 32 < Ksz) {
            wait_asynccnt0();      // our async writes to buf^1 are done
            __syncthreads();       // everyone done reading buf + writing buf^1
            buf ^= 1;
        }
    }

#pragma unroll
    for (int rh = 0; rh < 2; ++rh) {
#pragma unroll
        for (int nt = 0; nt < 4; ++nt) {
#pragma unroll
            for (int r = 0; r < 8; ++r) {
                const int row = m0b + mw * 32 + rh * 16 + r + 8 * hw;
                const int col = n0b + nw * 64 + nt * 16 + ln;
                const float v = acc[rh][nt][r];
                if (mode == 0) {
                    ((float*)Cout)[(size_t)row * Nsz + col] = v;
                } else {
                    const int b = row / Lc, l = row % Lc;
                    const int h = col / HDc, hd = col % HDc;
                    bf16_t* o = (bf16_t*)Cout;
                    if (mode == 1)
                        o[(((size_t)b * Hc + h) * Lc + l) * HDc + hd] = (bf16_t)v;
                    else if (mode == 2)
                        o[(((size_t)b * KVHc + h) * Lc + l) * HDc + hd] = (bf16_t)v;
                    else // V transposed: (b, kvh, hd, l)
                        o[(((size_t)b * KVHc + h) * HDc + hd) * Lc + l] = (bf16_t)v;
                }
            }
        }
    }
}

// ---------------------------------------------------------------------------
// Kernel 3: RMSNorm + partial RoPE + optional per-head gain, in place.
// One wave32 per 128-element head row.
// ---------------------------------------------------------------------------
__global__ __launch_bounds__(128)
void rms_rope(bf16_t* __restrict__ data, int nheads,
              const float* __restrict__ gain, int nrows) {
    const int row = blockIdx.x * 4 + (threadIdx.x >> 5);
    if (row >= nrows) return;
    const int lane = threadIdx.x & 31;
    const int l = row % Lc;
    const int h = (row / Lc) % nheads;
    bf16_t* p = data + (size_t)row * HDc;

    const int i0 = lane * 2;
    float a0 = (float)p[i0],      a1 = (float)p[i0 + 1];
    float b0 = (float)p[i0 + 64], b1 = (float)p[i0 + 65];

    float ss = a0 * a0 + a1 * a1 + b0 * b0 + b1 * b1;
#pragma unroll
    for (int m = 1; m < 32; m <<= 1) ss += __shfl_xor(ss, m, 32);
    const float rn = rsqrtf(ss * (1.0f / 128.0f) + EPSc);
    a0 *= rn; a1 *= rn; b0 *= rn; b1 *= rn;

    if (l >= Mc) {
        const float t = (float)(l - Mc);
        const float c = 0.14391156831212793f;   // ln(1e4)/64
        const float f0 = __expf(-(float)i0 * c);
        const float f1 = __expf(-(float)(i0 + 1) * c);
        float s0, c0, s1, c1;
        __sincosf(t * f0, &s0, &c0);
        __sincosf(t * f1, &s1, &c1);
        float na0 =  a0 * c0 + b0 * s0, nb0 = -a0 * s0 + b0 * c0;
        float na1 =  a1 * c1 + b1 * s1, nb1 = -a1 * s1 + b1 * c1;
        a0 = na0; b0 = nb0; a1 = na1; b1 = nb1;
    }
    if (gain) {
        const float g = gain[h];
        a0 *= g; a1 *= g; b0 *= g; b1 *= g;
    }
    p[i0]      = (bf16_t)a0; p[i0 + 1]  = (bf16_t)a1;
    p[i0 + 64] = (bf16_t)b0; p[i0 + 65] = (bf16_t)b1;
}

// ---------------------------------------------------------------------------
// Kernel 4: causal GQA flash attention. One wave per (b, h, 16-row q tile).
// ---------------------------------------------------------------------------
__global__ __launch_bounds__(256)
void attn_flash(const bf16_t* __restrict__ Qb, const bf16_t* __restrict__ Kb,
                const bf16_t* __restrict__ Vtb, bf16_t* __restrict__ Yb) {
    __shared__ __align__(16) bf16_t plds[8 * 16 * 32];   // 8 waves * 16x32 tile
    const int wib  = threadIdx.x >> 5;
    const int lane = threadIdx.x & 31;
    const int ln = lane & 15, hw = lane >> 4;

    const int wid = blockIdx.x * 8 + wib;
    const int qtiles = Lc / 16;
    const int qt = wid % qtiles;
    const int h  = (wid / qtiles) % Hc;
    const int b  = wid / (qtiles * Hc);
    const int l0 = qt * 16;
    const int kvh = h / (Hc / KVHc);

    const bf16_t* Qp = Qb + ((size_t)b * Hc + h) * Lc * HDc;
    const bf16_t* Kp = Kb + ((size_t)b * KVHc + kvh) * Lc * HDc;
    const bf16_t* Vp = Vtb + ((size_t)b * KVHc + kvh) * HDc * Lc;
    bf16_t* lp = &plds[wib * 512];

    v16bf qf[4];
    {
        const bf16_t* qr = Qp + (size_t)(l0 + ln) * HDc + hw * 8;
#pragma unroll
        for (int t = 0; t < 4; ++t)
            qf[t] = cat8(*(const v8bf*)(qr + t * 32),
                         *(const v8bf*)(qr + t * 32 + 16));
    }

    float mrow[8], lrow[8];
    v8f acc[8];
#pragma unroll
    for (int r = 0; r < 8; ++r) { mrow[r] = -__builtin_inff(); lrow[r] = 0.0f; }
#pragma unroll
    for (int t = 0; t < 8; ++t) acc[t] = zero_v8f();

    const float sc = 0.08838834764831845f;   // 1/sqrt(128)

    for (int j0 = 0; j0 < l0 + 16; j0 += 32) {
        v8f S0 = zero_v8f(), S1 = zero_v8f();
#pragma unroll
        for (int t = 0; t < 4; ++t) {
            const bf16_t* kp0 = Kp + (size_t)(j0 + ln) * HDc + t * 32 + hw * 8;
            const bf16_t* kp1 = Kp + (size_t)(j0 + 16 + ln) * HDc + t * 32 + hw * 8;
            v16bf kf0 = cat8(*(const v8bf*)kp0, *(const v8bf*)(kp0 + 16));
            v16bf kf1 = cat8(*(const v8bf*)kp1, *(const v8bf*)(kp1 + 16));
            S0 = wmma_bf16(qf[t], kf0, S0);
            S1 = wmma_bf16(qf[t], kf1, S1);
        }

        float alpha[8];
#pragma unroll
        for (int r = 0; r < 8; ++r) {
            const int rl = l0 + r + 8 * hw;
            float s0 = S0[r] * sc; if (j0 + ln > rl)      s0 = -__builtin_inff();
            float s1 = S1[r] * sc; if (j0 + 16 + ln > rl) s1 = -__builtin_inff();
            float rmax = fmaxf(s0, s1);
#pragma unroll
            for (int m = 1; m < 16; m <<= 1) rmax = fmaxf(rmax, __shfl_xor(rmax, m, 32));
            const float nm = fmaxf(mrow[r], rmax);
            const float al = __expf(mrow[r] - nm);
            const float p0 = __expf(s0 - nm);
            const float p1 = __expf(s1 - nm);
            float rs = p0 + p1;
#pragma unroll
            for (int m = 1; m < 16; m <<= 1) rs += __shfl_xor(rs, m, 32);
            lrow[r] = lrow[r] * al + rs;
            mrow[r] = nm;
            alpha[r] = al;
            lp[(r + 8 * hw) * 32 + ln]      = (bf16_t)p0;
            lp[(r + 8 * hw) * 32 + 16 + ln] = (bf16_t)p1;
        }
#pragma unroll
        for (int t = 0; t < 8; ++t)
#pragma unroll
            for (int r = 0; r < 8; ++r) acc[t][r] *= alpha[r];

        asm volatile("s_wait_dscnt 0x0" ::: "memory");
        v16bf pf = cat8(*(const v8bf*)(lp + ln * 32 + hw * 8),
                        *(const v8bf*)(lp + ln * 32 + hw * 8 + 16));

#pragma unroll
        for (int t = 0; t < 8; ++t) {
            const bf16_t* vp = Vp + (size_t)(t * 16 + ln) * Lc + j0 + hw * 8;
            v16bf vf = cat8(*(const v8bf*)vp, *(const v8bf*)(vp + 16));
            acc[t] = wmma_bf16(pf, vf, acc[t]);
        }
    }

    float inv[8];
#pragma unroll
    for (int r = 0; r < 8; ++r) inv[r] = 1.0f / lrow[r];
#pragma unroll
    for (int t = 0; t < 8; ++t)
#pragma unroll
        for (int r = 0; r < 8; ++r) {
            const int rl = l0 + r + 8 * hw;
            Yb[((size_t)b * Lc + rl) * Dc + h * HDc + t * 16 + ln] =
                (bf16_t)(acc[t][r] * inv[r]);
        }
}

// ---------------------------------------------------------------------------
// Host launch
// ---------------------------------------------------------------------------
extern "C" void kernel_launch(void* const* d_in, const int* in_sizes, int n_in,
                              void* d_out, int out_size, void* d_ws, size_t ws_size,
                              hipStream_t stream) {
    (void)in_sizes; (void)n_in; (void)out_size; (void)ws_size;
    const float* x  = (const float*)d_in[0];
    const float* Wq = (const float*)d_in[1];
    const float* Wk = (const float*)d_in[2];
    const float* Wv = (const float*)d_in[3];
    const float* Wp = (const float*)d_in[4];
    const float* qg = (const float*)d_in[5];

    char* ws = (char*)d_ws;
    size_t o = 0;
    auto take = [&](size_t bytes) { char* p = ws + o; o += (bytes + 255) & ~(size_t)255; return p; };
    bf16_t* xb  = (bf16_t*)take((size_t)BLc * Dc * 2);
    bf16_t* Wqb = (bf16_t*)take((size_t)Dc * Dc * 2);
    bf16_t* Wkb = (bf16_t*)take((size_t)KVD * Dc * 2);
    bf16_t* Wvb = (bf16_t*)take((size_t)KVD * Dc * 2);
    bf16_t* Wpb = (bf16_t*)take((size_t)Dc * Dc * 2);
    bf16_t* Qb  = (bf16_t*)take((size_t)Bc * Hc * Lc * HDc * 2);
    bf16_t* Kb  = (bf16_t*)take((size_t)Bc * KVHc * Lc * HDc * 2);
    bf16_t* Vtb = (bf16_t*)take((size_t)Bc * KVHc * HDc * Lc * 2);
    bf16_t* Yb  = (bf16_t*)take((size_t)BLc * Dc * 2);

    auto cast = [&](const float* src, bf16_t* dst, size_t n) {
        int n8 = (int)(n / 8);
        cast_f32_bf16<<<(n8 + 255) / 256, 256, 0, stream>>>(src, dst, n8);
    };
    cast(x,  xb,  (size_t)BLc * Dc);
    cast(Wq, Wqb, (size_t)Dc * Dc);
    cast(Wk, Wkb, (size_t)KVD * Dc);
    cast(Wv, Wvb, (size_t)KVD * Dc);
    cast(Wp, Wpb, (size_t)Dc * Dc);

    auto gemm = [&](const bf16_t* A, const bf16_t* Bw, void* C,
                    int Msz, int Nsz, int Ksz, int mode) {
        int blocks = (Msz / 128) * (Nsz / 128);
        gemm_bf16<<<blocks, 256, 0, stream>>>(A, Bw, C, Msz, Nsz, Ksz, mode);
    };
    gemm(xb, Wqb, Qb,  BLc, Dc,  Dc, 1);   // Q = x @ Wq^T  -> (b,h,l,hd)
    gemm(xb, Wkb, Kb,  BLc, KVD, Dc, 2);   // K = x @ Wk^T  -> (b,kvh,l,hd)
    gemm(xb, Wvb, Vtb, BLc, KVD, Dc, 3);   // V = x @ Wv^T  -> (b,kvh,hd,l)

    rms_rope<<<(Bc * Hc * Lc) / 4, 128, 0, stream>>>(Qb, Hc, qg, Bc * Hc * Lc);
    rms_rope<<<(Bc * KVHc * Lc) / 4, 128, 0, stream>>>(Kb, KVHc, nullptr, Bc * KVHc * Lc);

    attn_flash<<<(Bc * Hc * (Lc / 16)) / 8, 256, 0, stream>>>(Qb, Kb, Vtb, Yb);

    gemm(Yb, Wpb, d_out, BLc, Dc, Dc, 0);  // out = Y @ Wproj^T (f32)
}
